// SynGNNLayer_53738630808271
// MI455X (gfx1250) — matrix operand order repaired and checked by reading, not compiled, not run on hardware
//
#include <hip/hip_runtime.h>
#include <hip/hip_bf16.h>

// ---------------- problem constants (from reference setup_inputs) ----------
constexpr int N  = 16384;
constexpr int E  = 131072;
constexpr int D  = 128;
constexpr int H  = 4;
constexpr int De = 32;
constexpr int Dh = 2048;
constexpr int Do = 64;
constexpr int HD = H * D;        // 512
constexpr int EN = E + N;        // 147456 edges incl. self loops
constexpr float EPS = 1e-5f;
constexpr float NEG_SLOPE = 0.2f;

typedef __attribute__((ext_vector_type(16))) _Float16 v16h;
typedef __attribute__((ext_vector_type(8)))  _Float16 v8h;
typedef __attribute__((ext_vector_type(8)))  float    v8f;
typedef __attribute__((ext_vector_type(4)))  float    v4f;

static inline unsigned cdiv(long a, long b) { return (unsigned)((a + b - 1) / b); }

// ---------------- tiny utility kernels -------------------------------------
__global__ void fill_k(float* __restrict__ p, float v, long n) {
  long i = blockIdx.x * (long)blockDim.x + threadIdx.x;
  if (i < n) p[i] = v;
}

// fp32 -> fp16 conversion (4 elements/thread, 128-bit loads)
__global__ void cvt_f16_k(const float* __restrict__ in, _Float16* __restrict__ out, long n4) {
  long i = blockIdx.x * (long)blockDim.x + threadIdx.x;
  if (i >= n4) return;
  v4f v = *(const v4f*)(in + i * 4);
  _Float16* o = out + i * 4;
  o[0] = (_Float16)v.x; o[1] = (_Float16)v.y;
  o[2] = (_Float16)v.z; o[3] = (_Float16)v.w;
}

// global scalar reduction: sum and sum-of-squares
__global__ void reduce_sumsq_k(const float* __restrict__ x, long n, float* __restrict__ red) {
  __shared__ float s1[256], s2[256];
  float a = 0.f, b = 0.f;
  for (long i = blockIdx.x * (long)blockDim.x + threadIdx.x; i < n;
       i += (long)gridDim.x * blockDim.x) {
    float v = x[i];
    a += v; b += v * v;
  }
  s1[threadIdx.x] = a; s2[threadIdx.x] = b;
  __syncthreads();
  for (int s = 128; s > 0; s >>= 1) {
    if ((int)threadIdx.x < s) {
      s1[threadIdx.x] += s1[threadIdx.x + s];
      s2[threadIdx.x] += s2[threadIdx.x + s];
    }
    __syncthreads();
  }
  if (threadIdx.x == 0) { atomicAdd(&red[0], s1[0]); atomicAdd(&red[1], s2[0]); }
}

// graph_norm apply: scalar mean / population std over ALL entries, per-channel affine
__global__ void apply_norm_k(const float* __restrict__ x, const float* __restrict__ red,
                             const float* __restrict__ w, const float* __restrict__ b,
                             float* __restrict__ out, long n, int C) {
  long i = blockIdx.x * (long)blockDim.x + threadIdx.x;
  if (i >= n) return;
  float inv_n = 1.0f / (float)n;
  float mu  = red[0] * inv_n;
  float var = red[1] * inv_n - mu * mu;
  var = var > 0.f ? var : 0.f;
  float stdv = sqrtf(var);
  int c = (int)(i % C);
  out[i] = (x[i] - mu) / (stdv + EPS) * w[c] + b[c];
}

__global__ void add_k(const float* __restrict__ a, const float* __restrict__ b,
                      float* __restrict__ out, long n) {
  long i = blockIdx.x * (long)blockDim.x + threadIdx.x;
  if (i < n) out[i] = a[i] + b[i];
}

// ---------------- WMMA GEMM: C[M,Nc] = act(A[M,K] @ W[Nc,K]^T + bias) ------
// A and W are pre-converted fp16. One wave computes a 16x64 strip of C
// (4 n-tiles reuse one A fragment). Inner loop is pure 128-bit fp16 loads
// feeding v_wmma_f32_16x16x32_f16 (fp32 accumulate); no conversions.
template <bool RELU, bool OF16>
__global__ __launch_bounds__(256) void gemm_wmma_k(const _Float16* __restrict__ Ah,
                                                   const _Float16* __restrict__ Wh,
                                                   const float* __restrict__ bias,
                                                   void* __restrict__ Cp,
                                                   int M, int Nc, int K) {
  int wid = blockIdx.x * (blockDim.x >> 5) + (threadIdx.x >> 5);
  int groupsN = Nc >> 6;
  int total = (M >> 4) * groupsN;
  if (wid >= total) return;
  int tm  = wid / groupsN;
  int tn0 = (wid % groupsN) << 6;
  int lane = threadIdx.x & 31;
  int mrow = lane & 15;
  bool hi  = lane >= 16;

  v8f acc0 = {}, acc1 = {}, acc2 = {}, acc3 = {};
  long aRow = (long)(tm * 16 + mrow) * K;
  int kaB0 = hi ? 8 : 0;     // A frag elements 0..7  -> K = kaB0 + i
  int kaB1 = hi ? 24 : 16;   // A frag elements 8..15 -> K = kaB1 + (i-8)
  int kbB  = hi ? 8 : 0;     // B frag elements 0..7 -> K = kbB + i, 8..15 -> K = kbB + 16 + (i-8)

  for (int k0 = 0; k0 < K; k0 += 32) {
    v8h al = *(const v8h*)(Ah + aRow + k0 + kaB0);   // 16B-aligned contiguous runs
    v8h am = *(const v8h*)(Ah + aRow + k0 + kaB1);
    v16h a = __builtin_shufflevector(al, am, 0, 1, 2, 3, 4, 5, 6, 7,
                                     8, 9, 10, 11, 12, 13, 14, 15);
#pragma unroll
    for (int j = 0; j < 4; ++j) {
      int ncol = tn0 + j * 16 + mrow;     // B column held by this lane
      const _Float16* wp = Wh + (long)ncol * K + k0 + kbB;
      v8h b0 = *(const v8h*)(wp);
      v8h b1 = *(const v8h*)(wp + 16);
      v16h b = __builtin_shufflevector(b0, b1, 0, 1, 2, 3, 4, 5, 6, 7,
                                       8, 9, 10, 11, 12, 13, 14, 15);
      v8f& acc = (j == 0) ? acc0 : (j == 1) ? acc1 : (j == 2) ? acc2 : acc3;
      acc = __builtin_amdgcn_wmma_f32_16x16x32_f16(false, a, false, b,
                                                   (short)0, acc, false, false);
    }
  }
#pragma unroll
  for (int j = 0; j < 4; ++j) {
    int ncol = tn0 + j * 16 + mrow;
    float bs = bias ? bias[ncol] : 0.f;
    v8f acc = (j == 0) ? acc0 : (j == 1) ? acc1 : (j == 2) ? acc2 : acc3;
#pragma unroll
    for (int r = 0; r < 8; ++r) {
      int mr = tm * 16 + r + (hi ? 8 : 0);
      float v = acc[r] + bs;
      if (RELU) v = v > 0.f ? v : 0.f;
      if (OF16) ((_Float16*)Cp)[(long)mr * Nc + ncol] = (_Float16)v;
      else      ((float*)Cp)[(long)mr * Nc + ncol] = v;
    }
  }
}

// ---------------- edge-side kernels ----------------------------------------
// scatter-sum of edge_attr per dst + degree (for self-loop fill_value='mean')
__global__ void edge_easum_k(const int* __restrict__ ei, const float* __restrict__ ea,
                             float* __restrict__ easum, float* __restrict__ deg) {
  int t = blockIdx.x * blockDim.x + threadIdx.x;
  if (t >= E * De) return;
  int e = t >> 5, d = t & 31;
  int dn = ei[E + e];
  atomicAdd(&easum[dn * De + d], ea[t]);
  if (d == 0) atomicAdd(&deg[dn], 1.0f);
}

__global__ void ea_finalize_k(float* __restrict__ easum, const float* __restrict__ deg) {
  int t = blockIdx.x * blockDim.x + threadIdx.x;
  if (t >= N * De) return;
  float dg = deg[t >> 5];
  easum[t] /= (dg > 1.f ? dg : 1.f);
}

__device__ inline void atomicMaxFloat(float* addr, float val) {
  int old = __float_as_int(*addr);
  while (val > __int_as_float(old)) {
    int assumed = old;
    old = atomicCAS((int*)addr, assumed, __float_as_int(val));
    if (old == assumed) break;
  }
}

// one wave per edge: e_feat (ea @ We.T) on the fly, leaky-relu, dot with att,
// 8-lane shfl reduction per head, segment-max via CAS atomic
__global__ __launch_bounds__(256) void edge_score_k(const int* __restrict__ ei,
                                                    const float* __restrict__ ea,
                                                    const float* __restrict__ ea_mean,
                                                    const float* __restrict__ xl,
                                                    const float* __restrict__ xr,
                                                    const float* __restrict__ We,
                                                    const float* __restrict__ att,
                                                    float* __restrict__ score,
                                                    float* __restrict__ smax) {
  int wid = blockIdx.x * (blockDim.x >> 5) + (threadIdx.x >> 5);
  if (wid >= EN) return;
  int lane = threadIdx.x & 31;
  int sn, dn;
  const float* eaRow;
  if (wid < E) { sn = ei[wid]; dn = ei[E + wid]; eaRow = ea + (long)wid * De; }
  else         { sn = wid - E; dn = sn;          eaRow = ea_mean + (long)sn * De; }

  float eabuf[De];
#pragma unroll
  for (int d = 0; d < De; ++d) eabuf[d] = eaRow[d];

  int qbase = lane * 16;                 // 16 consecutive (h,c) channels per lane
  float partial = 0.f;
  const float* xls = xl + (long)sn * HD;
  const float* xrd = xr + (long)dn * HD;
#pragma unroll 4
  for (int j = 0; j < 16; ++j) {
    int q = qbase + j;
    const float* wr = We + (long)q * De;
    float ef = 0.f;
#pragma unroll
    for (int d = 0; d < De; ++d) ef += eabuf[d] * wr[d];
    float z = xls[q] + xrd[q] + ef;
    z = z > 0.f ? z : NEG_SLOPE * z;
    partial += att[q] * z;
  }
  // reduce over the 8 lanes belonging to one head (wave32, width-8 groups)
  for (int o = 4; o >= 1; o >>= 1) partial += __shfl_xor(partial, o, 8);
  if ((lane & 7) == 0) {
    int h = lane >> 3;
    score[(long)wid * H + h] = partial;
    atomicMaxFloat(&smax[dn * H + h], partial);
  }
}

__global__ void edge_exp_k(const int* __restrict__ ei, const float* __restrict__ score,
                           const float* __restrict__ smax, float* __restrict__ alpha,
                           float* __restrict__ ssum) {
  int t = blockIdx.x * blockDim.x + threadIdx.x;
  if (t >= EN * H) return;
  int e = t >> 2, h = t & 3;
  int dn = (e < E) ? ei[E + e] : (e - E);
  float ex = __expf(score[t] - smax[dn * H + h]);
  alpha[t] = ex;
  atomicAdd(&ssum[dn * H + h], ex);
}

__global__ void alpha_norm_k(const int* __restrict__ ei, float* __restrict__ alpha,
                             const float* __restrict__ ssum) {
  int t = blockIdx.x * blockDim.x + threadIdx.x;
  if (t >= EN * H) return;
  int e = t >> 2, h = t & 3;
  int dn = (e < E) ? ei[E + e] : (e - E);
  alpha[t] = alpha[t] / ssum[dn * H + h];
}

// one wave per edge: out_agg[dst] += x_l[src] * alpha
__global__ __launch_bounds__(256) void edge_agg_k(const int* __restrict__ ei,
                                                  const float* __restrict__ xl,
                                                  const float* __restrict__ alpha,
                                                  float* __restrict__ out_agg) {
  int wid = blockIdx.x * (blockDim.x >> 5) + (threadIdx.x >> 5);
  if (wid >= EN) return;
  int lane = threadIdx.x & 31;
  int sn, dn;
  if (wid < E) { sn = ei[wid]; dn = ei[E + wid]; }
  else         { sn = wid - E; dn = sn; }
  int h = lane >> 3;
  float al = alpha[(long)wid * H + h];
  int qbase = lane * 16;
  const float* xls = xl + (long)sn * HD + qbase;
  float* dsts = out_agg + (long)dn * HD + qbase;
#pragma unroll
  for (int j = 0; j < 16; ++j) atomicAdd(&dsts[j], xls[j] * al);
}

// head mean + gat_bias + residual with the normed input
__global__ void combine1_k(const float* __restrict__ xn0, const float* __restrict__ out_agg,
                           const float* __restrict__ gbias, float* __restrict__ out) {
  int i = blockIdx.x * blockDim.x + threadIdx.x;
  if (i >= N * D) return;
  int n = i >> 7, c = i & 127;
  const float* oa = out_agg + (long)n * HD;
  float m = 0.25f * (oa[c] + oa[128 + c] + oa[256 + c] + oa[384 + c]);
  out[i] = xn0[i] + m + gbias[c];
}

// ---------------- host-side orchestration ----------------------------------
extern "C" void kernel_launch(void* const* d_in, const int* in_sizes, int n_in,
                              void* d_out, int out_size, void* d_ws, size_t ws_size,
                              hipStream_t stream) {
  (void)in_sizes; (void)n_in; (void)out_size; (void)ws_size;
  const float* x    = (const float*)d_in[0];
  const int*   ei   = (const int*)d_in[1];
  const float* ea   = (const float*)d_in[2];
  const float* Wl   = (const float*)d_in[4];
  const float* bl   = (const float*)d_in[5];
  const float* Wr   = (const float*)d_in[6];
  const float* br   = (const float*)d_in[7];
  const float* We   = (const float*)d_in[8];
  const float* att  = (const float*)d_in[9];
  const float* gbias= (const float*)d_in[10];
  const float* W1   = (const float*)d_in[11];
  const float* b1   = (const float*)d_in[12];
  const float* W2   = (const float*)d_in[13];
  const float* b2   = (const float*)d_in[14];
  const float* Wc   = (const float*)d_in[15];
  const float* bc   = (const float*)d_in[16];
  const float* nw0  = (const float*)d_in[17];
  const float* nb0  = (const float*)d_in[18];
  const float* nw1  = (const float*)d_in[19];
  const float* nb1  = (const float*)d_in[20];
  const float* nw2  = (const float*)d_in[21];
  const float* nb2  = (const float*)d_in[22];
  const float* nw3  = (const float*)d_in[23];
  const float* nb3  = (const float*)d_in[24];

  // ---- workspace layout (float units; every block 16B aligned) ----
  float* ws = (float*)d_ws;
  size_t o = 0;
  float* xn0    = ws + o; o += (size_t)N * D;    // graph_norm(x)
  float* xl     = ws + o; o += (size_t)N * HD;   // x @ Wl.T   (later aliased by hid)
  float* xr     = ws + o; o += (size_t)N * HD;   // x @ Wr.T
  float* outagg = ws + o; o += (size_t)N * HD;   // GAT aggregation
  float* eamean = ws + o; o += (size_t)N * De;
  float* deg    = ws + o; o += (size_t)N;
  float* score  = ws + o; o += (size_t)EN * H;
  float* smax   = ws + o; o += (size_t)N * H;
  float* ssum   = ws + o; o += (size_t)N * H;
  float* bufA   = ws + o; o += (size_t)N * D;
  float* bufB   = ws + o; o += (size_t)N * D;
  float* bufC   = ws + o; o += (size_t)N * D;
  float* cls    = ws + o; o += (size_t)N * Do;
  float* red    = ws + o; o += 8;                // 4 (sum,sumsq) pairs
  // fp16 buffers (sized in halves, advance o by halves/2 floats)
  _Float16* Wlh  = (_Float16*)(ws + o); o += (size_t)HD * D  / 2;
  _Float16* Wrh  = (_Float16*)(ws + o); o += (size_t)HD * D  / 2;
  _Float16* W1h  = (_Float16*)(ws + o); o += (size_t)Dh * D  / 2;
  _Float16* W2h  = (_Float16*)(ws + o); o += (size_t)D  * Dh / 2;
  _Float16* Wch  = (_Float16*)(ws + o); o += (size_t)Do * D  / 2;
  _Float16* xn0h = (_Float16*)(ws + o); o += (size_t)N  * D  / 2;
  _Float16* acth = (_Float16*)(ws + o); o += (size_t)N  * D  / 2;
  // hid (f16, N x Dh = 67 MB) aliases xl+xr (exactly 2*N*HD floats), dead by then
  _Float16* hid = (_Float16*)xl;

  float* out_src   = (float*)d_out;              // [N, Do]
  float* out_alpha = out_src + (size_t)N * Do;   // [EN, H]

  const int T = 256;

  // ---- zero / init accumulators (every call: deterministic) ----
  fill_k<<<1, 8, 0, stream>>>(red, 0.f, 8);
  fill_k<<<cdiv(N, T), T, 0, stream>>>(deg, 0.f, N);
  fill_k<<<cdiv((long)N * De, T), T, 0, stream>>>(eamean, 0.f, (long)N * De);
  fill_k<<<cdiv((long)N * H, T), T, 0, stream>>>(ssum, 0.f, (long)N * H);
  fill_k<<<cdiv((long)N * H, T), T, 0, stream>>>(smax, -1e30f, (long)N * H);
  fill_k<<<cdiv((long)N * HD, T), T, 0, stream>>>(outagg, 0.f, (long)N * HD);

  // ---- one-time (per launch) fp16 weight conversions ----
  cvt_f16_k<<<cdiv((long)HD * D / 4, T), T, 0, stream>>>(Wl, Wlh, (long)HD * D / 4);
  cvt_f16_k<<<cdiv((long)HD * D / 4, T), T, 0, stream>>>(Wr, Wrh, (long)HD * D / 4);
  cvt_f16_k<<<cdiv((long)Dh * D / 4, T), T, 0, stream>>>(W1, W1h, (long)Dh * D / 4);
  cvt_f16_k<<<cdiv((long)D * Dh / 4, T), T, 0, stream>>>(W2, W2h, (long)D * Dh / 4);
  cvt_f16_k<<<cdiv((long)Do * D / 4, T), T, 0, stream>>>(Wc, Wch, (long)Do * D / 4);

  // ---- norm0: src = graph_norm(x) ----
  reduce_sumsq_k<<<1024, T, 0, stream>>>(x, (long)N * D, red + 0);
  apply_norm_k<<<cdiv((long)N * D, T), T, 0, stream>>>(x, red + 0, nw0, nb0, xn0,
                                                       (long)N * D, D);
  cvt_f16_k<<<cdiv((long)N * D / 4, T), T, 0, stream>>>(xn0, xn0h, (long)N * D / 4);

  // ---- GATv2 dense transforms via WMMA ----
  {
    unsigned waves = (N / 16) * (HD / 64);
    gemm_wmma_k<false, false><<<cdiv((long)waves * 32, T), T, 0, stream>>>(
        xn0h, Wlh, bl, xl, N, HD, D);
    gemm_wmma_k<false, false><<<cdiv((long)waves * 32, T), T, 0, stream>>>(
        xn0h, Wrh, br, xr, N, HD, D);
  }

  // ---- self-loop edge attr = scatter-mean ----
  edge_easum_k<<<cdiv((long)E * De, T), T, 0, stream>>>(ei, ea, eamean, deg);
  ea_finalize_k<<<cdiv((long)N * De, T), T, 0, stream>>>(eamean, deg);

  // ---- per-edge scores + segment softmax + aggregation ----
  unsigned edgeWaveBlocks = cdiv((long)EN * 32, T);
  edge_score_k<<<edgeWaveBlocks, T, 0, stream>>>(ei, ea, eamean, xl, xr, We, att,
                                                 score, smax);
  edge_exp_k<<<cdiv((long)EN * H, T), T, 0, stream>>>(ei, score, smax, out_alpha, ssum);
  alpha_norm_k<<<cdiv((long)EN * H, T), T, 0, stream>>>(ei, out_alpha, ssum);
  edge_agg_k<<<edgeWaveBlocks, T, 0, stream>>>(ei, xl, out_alpha, outagg);

  // ---- head mean + bias + residual ----
  combine1_k<<<cdiv((long)N * D, T), T, 0, stream>>>(xn0, outagg, gbias, bufA);

  // ---- norm1 ----
  reduce_sumsq_k<<<1024, T, 0, stream>>>(bufA, (long)N * D, red + 2);
  apply_norm_k<<<cdiv((long)N * D, T), T, 0, stream>>>(bufA, red + 2, nw1, nb1, bufB,
                                                       (long)N * D, D);
  cvt_f16_k<<<cdiv((long)N * D / 4, T), T, 0, stream>>>(bufB, acth, (long)N * D / 4);

  // ---- FFN: hid = relu(bufB @ W1.T + b1) [f16], bufC = hid @ W2.T + b2 ----
  {
    unsigned waves1 = (N / 16) * (Dh / 64);
    gemm_wmma_k<true, true><<<cdiv((long)waves1 * 32, T), T, 0, stream>>>(
        acth, W1h, b1, hid, N, Dh, D);
    unsigned waves2 = (N / 16) * (D / 64);
    gemm_wmma_k<false, false><<<cdiv((long)waves2 * 32, T), T, 0, stream>>>(
        hid, W2h, b2, bufC, N, D, Dh);
  }
  add_k<<<cdiv((long)N * D, T), T, 0, stream>>>(bufB, bufC, bufA, (long)N * D);

  // ---- norm2 ----
  reduce_sumsq_k<<<1024, T, 0, stream>>>(bufA, (long)N * D, red + 4);
  apply_norm_k<<<cdiv((long)N * D, T), T, 0, stream>>>(bufA, red + 4, nw2, nb2, bufB,
                                                       (long)N * D, D);
  cvt_f16_k<<<cdiv((long)N * D / 4, T), T, 0, stream>>>(bufB, acth, (long)N * D / 4);

  // ---- classifier GEMM ----
  {
    unsigned waves = (N / 16) * (Do / 64);
    gemm_wmma_k<false, false><<<cdiv((long)waves * 32, T), T, 0, stream>>>(
        acth, Wch, bc, cls, N, Do, D);
  }

  // ---- norm3 -> final output ----
  reduce_sumsq_k<<<1024, T, 0, stream>>>(cls, (long)N * Do, red + 6);
  apply_norm_k<<<cdiv((long)N * Do, T), T, 0, stream>>>(cls, red + 6, nw3, nb3, out_src,
                                                        (long)N * Do, Do);
}
